// EuclideanCodebook_9380208574523
// MI455X (gfx1250) — matrix operand order repaired and checked by reading, not compiled, not run on hardware
//
#include <hip/hip_runtime.h>

// ---------------------------------------------------------------------------
// EuclideanCodebook forward + EMA update for MI455X (gfx1250, wave32, WMMA)
//   B=16, T=2048, D=512, K=4096  ->  N = 32768 rows
// Distance GEMM via v_wmma_f32_16x16x32_bf16, split-bf16 (hi+lo) for
// ~f32-accurate argmin. Mtile=64 rows/WG (x-tile in 130KB dynamic LDS),
// each wave processes 2 column strips so every A fragment read from LDS
// feeds 6 WMMAs (24 wmma : 16 ds_load_b128 per d-step).
// ---------------------------------------------------------------------------

#define K_CODES 4096
#define DIMSZ   512
#define NROWS   32768
#define CB_EPS  1e-6f

#define MTILE    64              // rows per workgroup
#define A_STRIDE 520             // elements; 1040B row stride: 16B-aligned,
                                 // rows hit distinct 16B bank groups
#define A_BYTES  ((size_t)MTILE * A_STRIDE * 2)       // one hi or lo plane
#define SMEM_BYTES (2 * A_BYTES)                      // 133,120 bytes

typedef __attribute__((ext_vector_type(16))) __bf16         v16bf;
typedef __attribute__((ext_vector_type(8)))  float          v8f;
typedef __attribute__((ext_vector_type(16))) unsigned short us16;
typedef __attribute__((ext_vector_type(8)))  unsigned short us8;

static __device__ __forceinline__ unsigned short f32_to_bf16_rne(float f) {
    unsigned int u = __builtin_bit_cast(unsigned int, f);
    u += 0x7FFFu + ((u >> 16) & 1u);   // round-to-nearest-even (NaN ignored)
    return (unsigned short)(u >> 16);
}
static __device__ __forceinline__ float bf16_to_f32(unsigned short h) {
    unsigned int u = ((unsigned int)h) << 16;
    return __builtin_bit_cast(float, u);
}
static __device__ __forceinline__ us16 cat8(us8 a, us8 b) {
    us16 r;
#pragma unroll
    for (int i = 0; i < 8; ++i) { r[i] = a[i]; r[i + 8] = b[i]; }
    return r;
}

// ---------------------------------------------------------------------------
// Kernel 1: codebook -> bf16 hi/lo copies + ||e||^2 per code
// ---------------------------------------------------------------------------
__global__ void __launch_bounds__(256)
prep_embed_kernel(const float* __restrict__ embed,
                  unsigned short* __restrict__ ehi,
                  unsigned short* __restrict__ elo,
                  float* __restrict__ e2) {
    __shared__ float red[256];
    const int k = blockIdx.x;
    const float* row = embed + (size_t)k * DIMSZ;
    float acc = 0.f;
    for (int d = threadIdx.x; d < DIMSZ; d += 256) {
        float v = row[d];
        unsigned short h = f32_to_bf16_rne(v);
        ehi[(size_t)k * DIMSZ + d] = h;
        elo[(size_t)k * DIMSZ + d] = f32_to_bf16_rne(v - bf16_to_f32(h));
        acc += v * v;
    }
    red[threadIdx.x] = acc;
    __syncthreads();
    for (int s = 128; s > 0; s >>= 1) {
        if ((int)threadIdx.x < s) red[threadIdx.x] += red[threadIdx.x + s];
        __syncthreads();
    }
    if (threadIdx.x == 0) e2[k] = red[0];
}

// ---------------------------------------------------------------------------
// Kernel 2: fused  argmin_k( ||e_k||^2 - 2 * x_n . e_k )  via WMMA
//   block = 256 threads (8 waves), block owns 64 rows of x (4 row-tiles).
//   Each wave owns TWO 16-column strips per iteration (8 waves -> 256 cols),
//   reusing each A fragment across both strips and each B fragment across all 4 row-tiles.
// ---------------------------------------------------------------------------
__global__ void __launch_bounds__(256)
gemm_argmin_kernel(const float* __restrict__ x,
                   const unsigned short* __restrict__ ehi,
                   const unsigned short* __restrict__ elo,
                   const float* __restrict__ e2,
                   int* __restrict__ best_idx,
                   float* __restrict__ ind_out) {
    extern __shared__ char smem[];
    unsigned short (*Ahi)[A_STRIDE] = (unsigned short (*)[A_STRIDE])(smem);
    unsigned short (*Alo)[A_STRIDE] = (unsigned short (*)[A_STRIDE])(smem + A_BYTES);
    // reduction scratch aliases the A region (used only after column sweep)
    float* redV = (float*)(smem);                 // [8][64][16]
    int*   redI = (int*)(smem + 8 * 64 * 16 * 4); // [8][64][16]

    const int tid  = threadIdx.x;
    const int wave = tid >> 5;
    const int lane = tid & 31;
    const int lr    = lane & 15;   // A row within row-tile == B column in strip
    const int khalf = lane >> 4;   // which K-half this lane holds
    const int rowBase = blockIdx.x * MTILE;

    // ---- stage x tile (64 x 512 f32) into LDS as bf16 hi/lo ----
    for (int i = 0; i < MTILE * DIMSZ / 256; ++i) {   // 128 iters
        int off = i * 256 + tid;
        int r = off >> 9;
        int d = off & 511;
        float v = x[(size_t)(rowBase + r) * DIMSZ + d];
        unsigned short h = f32_to_bf16_rne(v);
        Ahi[r][d] = h;
        Alo[r][d] = f32_to_bf16_rne(v - bf16_to_f32(h));
    }
    __syncthreads();

    float bestVal[32];
    int   bestId[32];
#pragma unroll
    for (int v = 0; v < 32; ++v) { bestVal[v] = 3.4e38f; bestId[v] = 0; }

    for (int ct = 0; ct < K_CODES / 256; ++ct) {       // 16 iterations
        const int col0 = ct * 256 + wave * 16 + lr;    // strip 0 column
        const int col1 = col0 + 128;                   // strip 1 column
        const unsigned short* bh0 = ehi + (size_t)col0 * DIMSZ + khalf * 16;
        const unsigned short* bl0 = elo + (size_t)col0 * DIMSZ + khalf * 16;
        const unsigned short* bh1 = ehi + (size_t)col1 * DIMSZ + khalf * 16;
        const unsigned short* bl1 = elo + (size_t)col1 * DIMSZ + khalf * 16;

        v8f acc[8];                                    // [strip*4 + rt]
#pragma unroll
        for (int a = 0; a < 8; ++a) acc[a] = (v8f){};

#pragma unroll 2
        for (int d0 = 0; d0 < DIMSZ; d0 += 32) {
            // B fragments (this code's k order, matched by A): 16 contiguous K at khalf*16
            v16bf b_hi0 = __builtin_bit_cast(v16bf, *(const us16*)(bh0 + d0));
            v16bf b_lo0 = __builtin_bit_cast(v16bf, *(const us16*)(bl0 + d0));
            v16bf b_hi1 = __builtin_bit_cast(v16bf, *(const us16*)(bh1 + d0));
            v16bf b_lo1 = __builtin_bit_cast(v16bf, *(const us16*)(bl1 + d0));
#pragma unroll
            for (int rt = 0; rt < 4; ++rt) {
                const int r = rt * 16 + lr;
                // A fragment (contiguous, matches B): K = d0+khalf*16+[0..15]
                us8 a0h = *(const us8*)&Ahi[r][d0 + khalf * 16];
                us8 a1h = *(const us8*)&Ahi[r][d0 + 8 + khalf * 16];
                us8 a0l = *(const us8*)&Alo[r][d0 + khalf * 16];
                us8 a1l = *(const us8*)&Alo[r][d0 + 8 + khalf * 16];
                v16bf a_hi = __builtin_bit_cast(v16bf, cat8(a0h, a1h));
                v16bf a_lo = __builtin_bit_cast(v16bf, cat8(a0l, a1l));
                acc[rt] = __builtin_amdgcn_wmma_f32_16x16x32_bf16(
                              false, a_hi, false, b_hi0, (short)0, acc[rt], false, false);
                acc[rt] = __builtin_amdgcn_wmma_f32_16x16x32_bf16(
                              false, a_hi, false, b_lo0, (short)0, acc[rt], false, false);
                acc[rt] = __builtin_amdgcn_wmma_f32_16x16x32_bf16(
                              false, a_lo, false, b_hi0, (short)0, acc[rt], false, false);
                acc[4 + rt] = __builtin_amdgcn_wmma_f32_16x16x32_bf16(
                              false, a_hi, false, b_hi1, (short)0, acc[4 + rt], false, false);
                acc[4 + rt] = __builtin_amdgcn_wmma_f32_16x16x32_bf16(
                              false, a_hi, false, b_lo1, (short)0, acc[4 + rt], false, false);
                acc[4 + rt] = __builtin_amdgcn_wmma_f32_16x16x32_bf16(
                              false, a_lo, false, b_hi1, (short)0, acc[4 + rt], false, false);
            }
        }

        // C layout: lane holds column (lane&15); VGPR v holds row v + 8*khalf
        const float e2c0 = e2[col0];
        const float e2c1 = e2[col1];
#pragma unroll
        for (int rt = 0; rt < 4; ++rt) {
#pragma unroll
            for (int v = 0; v < 8; ++v) {
                const int b = rt * 8 + v;
                float d0s = e2c0 - 2.0f * acc[rt][v];       // strip 0 first
                if (d0s < bestVal[b]) { bestVal[b] = d0s; bestId[b] = col0; }
                float d1s = e2c1 - 2.0f * acc[4 + rt][v];   // then strip 1
                if (d1s < bestVal[b]) { bestVal[b] = d1s; bestId[b] = col1; }
            }
        }
    }

    // ---- cross-lane / cross-wave argmin reduction (aliases A region) ----
    __syncthreads();   // everyone done reading A before we overwrite it
#pragma unroll
    for (int rt = 0; rt < 4; ++rt) {
#pragma unroll
        for (int v = 0; v < 8; ++v) {
            int row64 = rt * 16 + v + 8 * khalf;
            redV[(wave * 64 + row64) * 16 + lr] = bestVal[rt * 8 + v];
            redI[(wave * 64 + row64) * 16 + lr] = bestId[rt * 8 + v];
        }
    }
    __syncthreads();
    if (tid < MTILE) {
        const int row = tid;
        float bv = 3.4e38f; int bi = 0;
        for (int w = 0; w < 8; ++w)
            for (int c = 0; c < 16; ++c) {
                float val = redV[(w * 64 + row) * 16 + c];
                int   id  = redI[(w * 64 + row) * 16 + c];
                if (val < bv || (val == bv && id < bi)) { bv = val; bi = id; }
            }
        best_idx[rowBase + row] = bi;
        ind_out[rowBase + row]  = (float)bi;
    }
}

// ---------------------------------------------------------------------------
// Kernel 3: quantize gather (OLD embed) + scatter-add stats
// ---------------------------------------------------------------------------
__global__ void __launch_bounds__(256)
gather_scatter_kernel(const float* __restrict__ x,
                      const float* __restrict__ embed,
                      const int* __restrict__ best_idx,
                      float* __restrict__ quant_out,
                      float* __restrict__ embed_sum,
                      float* __restrict__ counts) {
    const int n = blockIdx.x;
    const int idx = best_idx[n];
    const float* erow = embed     + (size_t)idx * DIMSZ;
    const float* xrow = x         + (size_t)n   * DIMSZ;
    float*       qrow = quant_out + (size_t)n   * DIMSZ;
    float*       srow = embed_sum + (size_t)idx * DIMSZ;
    for (int d = threadIdx.x; d < DIMSZ; d += 256) {
        qrow[d] = erow[d];
        atomicAdd(&srow[d], xrow[d]);
    }
    if (threadIdx.x == 0) atomicAdd(&counts[idx], 1.0f);
}

// ---------------------------------------------------------------------------
// Kernel 4: new_cluster_size = cs*decay + counts*(1-decay); total = sum
// ---------------------------------------------------------------------------
__global__ void __launch_bounds__(256)
cluster_kernel(const float* __restrict__ cluster_size,
               const float* __restrict__ counts,
               const float* __restrict__ decay_p,
               float* __restrict__ ncs_out,
               float* __restrict__ total) {
    __shared__ float red[256];
    const int k = blockIdx.x * 256 + threadIdx.x;
    const float decay = *decay_p;
    float v = cluster_size[k] * decay + counts[k] * (1.f - decay);
    ncs_out[k] = v;
    red[threadIdx.x] = v;
    __syncthreads();
    for (int s = 128; s > 0; s >>= 1) {
        if ((int)threadIdx.x < s) red[threadIdx.x] += red[threadIdx.x + s];
        __syncthreads();
    }
    if (threadIdx.x == 0) atomicAdd(total, red[0]);
}

// ---------------------------------------------------------------------------
// Kernel 5: new_embed_avg + laplace-smoothed new_embed
// ---------------------------------------------------------------------------
__global__ void __launch_bounds__(256)
final_kernel(const float* __restrict__ embed_avg,
             const float* __restrict__ embed_sum,
             const float* __restrict__ ncs,
             const float* __restrict__ total,
             const float* __restrict__ decay_p,
             float* __restrict__ nea_out,
             float* __restrict__ ne_out) {
    const size_t i = (size_t)blockIdx.x * 256 + threadIdx.x;
    const int k = (int)(i >> 9);                 // i / DIMSZ
    const float decay = *decay_p;
    float nea = embed_avg[i] * decay + embed_sum[i] * (1.f - decay);
    nea_out[i] = nea;
    const float tot = *total;
    const float sm = (ncs[k] + CB_EPS) / (tot + CB_EPS * (float)K_CODES) * tot;
    ne_out[i] = nea / sm;
}

// ---------------------------------------------------------------------------
extern "C" void kernel_launch(void* const* d_in, const int* in_sizes, int n_in,
                              void* d_out, int out_size, void* d_ws, size_t ws_size,
                              hipStream_t stream) {
    (void)in_sizes; (void)n_in; (void)out_size; (void)ws_size;

    const float* x          = (const float*)d_in[0];
    const float* embed      = (const float*)d_in[1];
    const float* cluster_sz = (const float*)d_in[2];
    const float* embed_avg  = (const float*)d_in[3];
    const float* decay      = (const float*)d_in[4];

    char* ws = (char*)d_ws;
    unsigned short* ehi = (unsigned short*)ws; ws += (size_t)K_CODES * DIMSZ * 2;
    unsigned short* elo = (unsigned short*)ws; ws += (size_t)K_CODES * DIMSZ * 2;
    float* e2        = (float*)ws;  ws += (size_t)K_CODES * 4;
    float* embed_sum = (float*)ws;  ws += (size_t)K_CODES * DIMSZ * 4;
    float* counts    = (float*)ws;  ws += (size_t)K_CODES * 4;
    int*   bestIdx   = (int*)ws;    ws += (size_t)NROWS * 4;
    float* total     = (float*)ws;  ws += 256;

    float* out     = (float*)d_out;
    float* q_out   = out;                              // [N, D]
    float* ind_out = q_out + (size_t)NROWS * DIMSZ;    // [N]
    float* ncs_out = ind_out + NROWS;                  // [K]
    float* nea_out = ncs_out + K_CODES;                // [K, D]
    float* ne_out  = nea_out + (size_t)K_CODES * DIMSZ;// [K, D]

    hipMemsetAsync(embed_sum, 0, (size_t)K_CODES * DIMSZ * 4, stream);
    hipMemsetAsync(counts,    0, (size_t)K_CODES * 4,         stream);
    hipMemsetAsync(total,     0, 4,                           stream);

    prep_embed_kernel<<<K_CODES, 256, 0, stream>>>(embed, ehi, elo, e2);
    gemm_argmin_kernel<<<NROWS / MTILE, 256, SMEM_BYTES, stream>>>(
        x, ehi, elo, e2, bestIdx, ind_out);
    gather_scatter_kernel<<<NROWS, 256, 0, stream>>>(x, embed, bestIdx,
                                                     q_out, embed_sum, counts);
    cluster_kernel<<<K_CODES / 256, 256, 0, stream>>>(cluster_sz, counts, decay,
                                                      ncs_out, total);
    final_kernel<<<(K_CODES * DIMSZ) / 256, 256, 0, stream>>>(
        embed_avg, embed_sum, ncs_out, total, decay, nea_out, ne_out);
}
